// EdgeConv_17875653886624
// MI455X (gfx1250) — matrix operand order repaired and checked by reading, not compiled, not run on hardware
//
#include <hip/hip_runtime.h>
#include <hip/hip_bf16.h>

// ---------------------------------------------------------------------------
// EdgeConv (DGCNN) for MI455X / gfx1250, wave32 + WMMA bf16.
// Pipeline: zero -> knn -> pack_x(bf16) -> gemm1(wmma) -> bn1 -> gemm2(wmma)
//           -> bn2 -> scatter-max.
// Fixed problem shape from reference: B=8, C=64, O=64, KNN=20.
// ---------------------------------------------------------------------------

#define KNN 20
#define TPW 4   // tiles (of 16 edges) per wave in the GEMM kernels

typedef __attribute__((ext_vector_type(8)))  float        v8f;
typedef __attribute__((ext_vector_type(8)))  unsigned int v8u;
typedef __attribute__((ext_vector_type(16))) __bf16       v16bf;

static __device__ __forceinline__ v16bf as_bf16(v8u u) {
    union { v8u u; v16bf b; } x; x.u = u; return x.b;
}
// f32 -> bf16 round-to-nearest-even, packed pair into one dword
static __device__ __forceinline__ unsigned f2bf(float f) {
    unsigned u = __float_as_uint(f);
    unsigned r = u + 0x7FFFu + ((u >> 16) & 1u);
    return r >> 16;
}
static __device__ __forceinline__ unsigned pack2(float lo, float hi) {
    return f2bf(lo) | (f2bf(hi) << 16);
}
static __device__ __forceinline__ float bflo(unsigned u) { return __uint_as_float(u << 16); }
static __device__ __forceinline__ float bfhi(unsigned u) { return __uint_as_float(u & 0xFFFF0000u); }
// (bf16(a)-bf16(b)) per half, repacked
static __device__ __forceinline__ unsigned subpack(unsigned a, unsigned b) {
    return pack2(bflo(a) - bflo(b), bfhi(a) - bfhi(b));
}

// ---------------------------------------------------------------------------
__global__ void zero_f32_kernel(float* __restrict__ p, long n) {
    long i = (long)blockIdx.x * 256 + threadIdx.x;
    if (i < n) p[i] = 0.0f;
}

// ---------------------------------------------------------------------------
// kNN per cloud: block covers 256 query points of one cloud; cloud positions
// staged in LDS; register top-20 (smallest d^2), fully unrolled insertion.
__global__ __launch_bounds__(256) void knn_kernel(const float* __restrict__ pos,
                                                  int* __restrict__ nbr, int P) {
    __shared__ float sp[2048 * 3];
    int cblocks = P >> 8;                       // 256-query blocks per cloud
    int cloud   = blockIdx.x / cblocks;
    int qlocal  = (blockIdx.x % cblocks) * 256 + threadIdx.x;
    const float* cp = pos + (size_t)cloud * P * 3;
    for (int idx = threadIdx.x; idx < P * 3; idx += 256) sp[idx] = cp[idx];
    __syncthreads();

    float qx = sp[qlocal * 3 + 0], qy = sp[qlocal * 3 + 1], qz = sp[qlocal * 3 + 2];
    float bd[KNN]; int bi[KNN];
#pragma unroll
    for (int t = 0; t < KNN; ++t) { bd[t] = 3.0e38f; bi[t] = 0; }

    for (int j = 0; j < P; ++j) {
        float dx = sp[j * 3 + 0] - qx;
        float dy = sp[j * 3 + 1] - qy;
        float dz = sp[j * 3 + 2] - qz;
        float d2 = dx * dx + dy * dy + dz * dz;
        if (j == qlocal) continue;              // no self-loops
        if (d2 < bd[KNN - 1]) {
            float cd = d2; int ci = j;
#pragma unroll
            for (int t = 0; t < KNN; ++t) {     // bubble into sorted position
                if (cd < bd[t]) {
                    float td = bd[t]; bd[t] = cd; cd = td;
                    int   ti = bi[t]; bi[t] = ci; ci = ti;
                }
            }
        }
    }
    int q = cloud * P + qlocal;
#pragma unroll
    for (int t = 0; t < KNN; ++t) nbr[q * KNN + t] = cloud * P + bi[t];
}

// ---------------------------------------------------------------------------
// Pack x (N x 64 f32) into bf16 pairs: xb[n*32 + p] = {x[n][2p], x[n][2p+1]}
__global__ void pack_x_kernel(const float* __restrict__ x,
                              unsigned* __restrict__ xb, int total) {
    int idx = blockIdx.x * 256 + threadIdx.x;
    if (idx >= total) return;
    float2 v = ((const float2*)x)[idx];
    xb[idx] = pack2(v.x, v.y);
}

// ---------------------------------------------------------------------------
// GEMM1: h1[e, :64] = [x_j | x_i - x_j] @ W1^T + b1, fused gather + bf16 WMMA,
// plus per-channel sum / sumsq for BatchNorm statistics.
// Wave tile = 16 edges x 64 channels: 4 K-chunks x 4 N-tiles = 16 WMMAs.
__global__ __launch_bounds__(256) void gemm1_kernel(
    const unsigned* __restrict__ xb, const int* __restrict__ nbr,
    const float* __restrict__ W1, const float* __restrict__ b1,
    float* __restrict__ h1, float* __restrict__ gsum, float* __restrict__ gsq,
    int E) {
    __shared__ unsigned wlds[64][65];   // row n, packed K-pair p  (pad -> no bank conflict)
    __shared__ float rsum[64], rsq[64];

    int tid = threadIdx.x;
    for (int idx = tid; idx < 64 * 64; idx += 256) {    // W1: 64 x 128
        int n = idx >> 6, p = idx & 63;
        wlds[n][p] = pack2(W1[n * 128 + 2 * p], W1[n * 128 + 2 * p + 1]);
    }
    if (tid < 64) { rsum[tid] = 0.f; rsq[tid] = 0.f; }
    __syncthreads();

    int wave = tid >> 5, lane = tid & 31;
    int m = lane & 15, hi = lane >> 4;                  // A: M=m ; B/C/D: N=m
    float bias[4];
#pragma unroll
    for (int nt = 0; nt < 4; ++nt) bias[nt] = b1[nt * 16 + m];

    float s[4] = {0.f, 0.f, 0.f, 0.f}, s2[4] = {0.f, 0.f, 0.f, 0.f};
    int tiles = E >> 4;
    int tbase = (blockIdx.x * 8 + wave) * TPW;
    v8f zacc = {0.f, 0.f, 0.f, 0.f, 0.f, 0.f, 0.f, 0.f};

    for (int t = 0; t < TPW; ++t) {
        int tile = tbase + t;
        if (tile >= tiles) break;
        int e = tile * 16 + m;
        int i = e / KNN;                                // center (col)
        int j = nbr[e];                                 // neighbor (row)
        const unsigned* xi = xb + (size_t)i * 32;
        const unsigned* xj = xb + (size_t)j * 32;

        v8f acc[4] = {zacc, zacc, zacc, zacc};
#pragma unroll
        for (int ck = 0; ck < 4; ++ck) {
            // A fragment (16x32 bf16): lane pairs K = 2v+8hi (v<4), 16+2(v-4)+8hi (v>=4)
            v8u au;
            if (ck < 2) {                               // features 0..63: x_j
                uint4 l = *(const uint4*)(xj + ck * 16 + 4 * hi);
                uint4 h = *(const uint4*)(xj + ck * 16 + 8 + 4 * hi);
                au[0] = l.x; au[1] = l.y; au[2] = l.z; au[3] = l.w;
                au[4] = h.x; au[5] = h.y; au[6] = h.z; au[7] = h.w;
            } else {                                    // features 64..127: x_i - x_j
                int ko = (ck - 2) * 16;
                uint4 jl = *(const uint4*)(xj + ko + 4 * hi);
                uint4 jh = *(const uint4*)(xj + ko + 8 + 4 * hi);
                uint4 il = *(const uint4*)(xi + ko + 4 * hi);
                uint4 ih = *(const uint4*)(xi + ko + 8 + 4 * hi);
                au[0] = subpack(il.x, jl.x); au[1] = subpack(il.y, jl.y);
                au[2] = subpack(il.z, jl.z); au[3] = subpack(il.w, jl.w);
                au[4] = subpack(ih.x, jh.x); au[5] = subpack(ih.y, jh.y);
                au[6] = subpack(ih.z, jh.z); au[7] = subpack(ih.w, jh.w);
            }
            v16bf af = as_bf16(au);
#pragma unroll
            for (int nt = 0; nt < 4; ++nt) {
                // B fragment (32x16 bf16): lane n=m, pair index = ck*16 + 4*hi + (v&3) + 8*(v>>2)
                v8u bu;
                int pb = ck * 16 + 4 * hi;
                int rn = nt * 16 + m;
#pragma unroll
                for (int v = 0; v < 8; ++v) bu[v] = wlds[rn][pb + (v & 3) + 8 * (v >> 2)];
                acc[nt] = __builtin_amdgcn_wmma_f32_16x16x32_bf16(
                    false, af, false, as_bf16(bu), (short)0, acc[nt], false, false);
            }
        }
        // D layout: VGPR r -> edge row M = r + 8*hi, channel N = m
#pragma unroll
        for (int nt = 0; nt < 4; ++nt) {
            int c = nt * 16 + m;
#pragma unroll
            for (int r = 0; r < 8; ++r) {
                float v = acc[nt][r] + bias[nt];
                int er = tile * 16 + r + 8 * hi;
                h1[(size_t)er * 64 + c] = v;
                s[nt] += v; s2[nt] += v * v;
            }
        }
    }
#pragma unroll
    for (int nt = 0; nt < 4; ++nt) {
        int c = nt * 16 + m;
        atomicAdd(&rsum[c], s[nt]);
        atomicAdd(&rsq[c], s2[nt]);
    }
    __syncthreads();
    if (tid < 64) { atomicAdd(&gsum[tid], rsum[tid]); atomicAdd(&gsq[tid], rsq[tid]); }
}

// ---------------------------------------------------------------------------
// Fold BN into y = h*a + c:  a = g*rsqrt(var+eps), c = beta - mu*a
__global__ void bn_finalize_kernel(const float* __restrict__ sum,
                                   const float* __restrict__ sq,
                                   const float* __restrict__ g,
                                   const float* __restrict__ beta,
                                   float* __restrict__ a, float* __restrict__ c,
                                   float invE) {
    int t = threadIdx.x;
    float mu  = sum[t] * invE;
    float var = sq[t] * invE - mu * mu;
    float inv = rsqrtf(var + 1e-5f);
    float ga  = g[t] * inv;
    a[t] = ga;
    c[t] = beta[t] - mu * ga;
}

// ---------------------------------------------------------------------------
// GEMM2: h2 = relu(bn1(h1)) @ W2^T + b2, BN1+ReLU fused into bf16 operand
// build; per-channel BN2 statistics accumulated as in GEMM1.
__global__ __launch_bounds__(256) void gemm2_kernel(
    const float* __restrict__ h1, const float* __restrict__ W2,
    const float* __restrict__ b2, const float* __restrict__ a1,
    const float* __restrict__ c1, float* __restrict__ h2,
    float* __restrict__ gsum, float* __restrict__ gsq, int E) {
    __shared__ unsigned wlds[64][33];   // W2: 64 x 64 -> 32 packed pairs/row
    __shared__ float sa[64], sc[64], rsum[64], rsq[64];

    int tid = threadIdx.x;
    for (int idx = tid; idx < 64 * 32; idx += 256) {
        int n = idx >> 5, p = idx & 31;
        wlds[n][p] = pack2(W2[n * 64 + 2 * p], W2[n * 64 + 2 * p + 1]);
    }
    if (tid < 64) { sa[tid] = a1[tid]; sc[tid] = c1[tid]; rsum[tid] = 0.f; rsq[tid] = 0.f; }
    __syncthreads();

    int wave = tid >> 5, lane = tid & 31;
    int m = lane & 15, hi = lane >> 4;
    float bias[4];
#pragma unroll
    for (int nt = 0; nt < 4; ++nt) bias[nt] = b2[nt * 16 + m];

    float s[4] = {0.f, 0.f, 0.f, 0.f}, s2[4] = {0.f, 0.f, 0.f, 0.f};
    int tiles = E >> 4;
    int tbase = (blockIdx.x * 8 + wave) * TPW;
    v8f zacc = {0.f, 0.f, 0.f, 0.f, 0.f, 0.f, 0.f, 0.f};

    for (int t = 0; t < TPW; ++t) {
        int tile = tbase + t;
        if (tile >= tiles) break;
        int e = tile * 16 + m;
        const float* row = h1 + (size_t)e * 64;

        v8f acc[4] = {zacc, zacc, zacc, zacc};
#pragma unroll
        for (int ck = 0; ck < 2; ++ck) {
            int base = ck * 32 + 8 * hi;
            float4 q0 = *(const float4*)(row + base);
            float4 q1 = *(const float4*)(row + base + 4);
            float4 q2 = *(const float4*)(row + base + 16);
            float4 q3 = *(const float4*)(row + base + 20);
            // bn1 + relu per element (channel = K index)
            #define BNR(v, k) fmaxf((v) * sa[k] + sc[k], 0.f)
            v8u au;
            au[0] = pack2(BNR(q0.x, base + 0),  BNR(q0.y, base + 1));
            au[1] = pack2(BNR(q0.z, base + 2),  BNR(q0.w, base + 3));
            au[2] = pack2(BNR(q1.x, base + 4),  BNR(q1.y, base + 5));
            au[3] = pack2(BNR(q1.z, base + 6),  BNR(q1.w, base + 7));
            au[4] = pack2(BNR(q2.x, base + 16), BNR(q2.y, base + 17));
            au[5] = pack2(BNR(q2.z, base + 18), BNR(q2.w, base + 19));
            au[6] = pack2(BNR(q3.x, base + 20), BNR(q3.y, base + 21));
            au[7] = pack2(BNR(q3.z, base + 22), BNR(q3.w, base + 23));
            #undef BNR
            v16bf af = as_bf16(au);
#pragma unroll
            for (int nt = 0; nt < 4; ++nt) {
                v8u bu;
                int pb = ck * 16 + 4 * hi;
                int rn = nt * 16 + m;
#pragma unroll
                for (int v = 0; v < 8; ++v) bu[v] = wlds[rn][pb + (v & 3) + 8 * (v >> 2)];
                acc[nt] = __builtin_amdgcn_wmma_f32_16x16x32_bf16(
                    false, af, false, as_bf16(bu), (short)0, acc[nt], false, false);
            }
        }
#pragma unroll
        for (int nt = 0; nt < 4; ++nt) {
            int c = nt * 16 + m;
#pragma unroll
            for (int r = 0; r < 8; ++r) {
                float v = acc[nt][r] + bias[nt];
                int er = tile * 16 + r + 8 * hi;
                h2[(size_t)er * 64 + c] = v;
                s[nt] += v; s2[nt] += v * v;
            }
        }
    }
#pragma unroll
    for (int nt = 0; nt < 4; ++nt) {
        int c = nt * 16 + m;
        atomicAdd(&rsum[c], s[nt]);
        atomicAdd(&rsq[c], s2[nt]);
    }
    __syncthreads();
    if (tid < 64) { atomicAdd(&gsum[tid], rsum[tid]); atomicAdd(&gsq[tid], rsq[tid]); }
}

// ---------------------------------------------------------------------------
// Final: apply bn2+relu to h2 and scatter-max by row (neighbor). Values are
// >= 0 post-ReLU, so uint atomicMax on float bits with zero-init is exact
// (and matches the reference's "-inf -> 0" empty-row rule).
__global__ __launch_bounds__(256) void scatter_max_kernel(
    const float* __restrict__ h2, const int* __restrict__ nbr,
    const float* __restrict__ a2, const float* __restrict__ c2,
    unsigned* __restrict__ out, int E) {
    int e = blockIdx.x * 256 + threadIdx.x;
    if (e >= E) return;
    int r = nbr[e];
    const float* row = h2 + (size_t)e * 64;
    unsigned* orow = out + (size_t)r * 64;
#pragma unroll 4
    for (int c4 = 0; c4 < 16; ++c4) {
        float4 v = ((const float4*)row)[c4];
        float4 A = ((const float4*)a2)[c4];
        float4 C = ((const float4*)c2)[c4];
        atomicMax(orow + c4 * 4 + 0, __float_as_uint(fmaxf(v.x * A.x + C.x, 0.f)));
        atomicMax(orow + c4 * 4 + 1, __float_as_uint(fmaxf(v.y * A.y + C.y, 0.f)));
        atomicMax(orow + c4 * 4 + 2, __float_as_uint(fmaxf(v.z * A.z + C.z, 0.f)));
        atomicMax(orow + c4 * 4 + 3, __float_as_uint(fmaxf(v.w * A.w + C.w, 0.f)));
    }
}

// ---------------------------------------------------------------------------
extern "C" void kernel_launch(void* const* d_in, const int* in_sizes, int n_in,
                              void* d_out, int out_size, void* d_ws, size_t ws_size,
                              hipStream_t stream) {
    const float* x     = (const float*)d_in[0];
    const float* pos   = (const float*)d_in[1];
    const float* W1    = (const float*)d_in[3];
    const float* b1    = (const float*)d_in[4];
    const float* g1    = (const float*)d_in[5];
    const float* beta1 = (const float*)d_in[6];
    const float* W2    = (const float*)d_in[7];
    const float* b2    = (const float*)d_in[8];
    const float* g2    = (const float*)d_in[9];
    const float* beta2 = (const float*)d_in[10];

    const int N = in_sizes[0] / 64;   // points
    const int P = N / 8;              // points per cloud (B = 8)
    const int E = N * KNN;            // edges

    // workspace layout
    char* ws = (char*)d_ws;
    size_t off = 0;
    int*      nbr = (int*)(ws + off);      off += (size_t)E * 4;
    unsigned* xb  = (unsigned*)(ws + off); off += (size_t)N * 32 * 4;
    float*    h1  = (float*)(ws + off);    off += (size_t)E * 64 * 4;
    float*    h2  = (float*)(ws + off);    off += (size_t)E * 64 * 4;
    float*    stats = (float*)(ws + off);  // 8 x 64 floats
    float* sum1 = stats;       float* sq1 = stats + 64;
    float* sum2 = stats + 128; float* sq2 = stats + 192;
    float* a1c  = stats + 256; float* c1c = stats + 320;
    float* a2c  = stats + 384; float* c2c = stats + 448;

    const float invE = 1.0f / (float)E;

    // 1) zero output + BN stat accumulators
    long outN = (long)N * 64;
    zero_f32_kernel<<<(int)((outN + 255) / 256), 256, 0, stream>>>((float*)d_out, outN);
    zero_f32_kernel<<<1, 256, 0, stream>>>(stats, 256);

    // 2) kNN graph
    knn_kernel<<<N / 256, 256, 0, stream>>>(pos, nbr, P);

    // 3) pack x to bf16 pairs
    int packTot = N * 32;
    pack_x_kernel<<<(packTot + 255) / 256, 256, 0, stream>>>(x, xb, packTot);

    // 4) layer 1 GEMM (WMMA) + stats
    int tiles = E / 16;
    int gblocks = (tiles + 8 * TPW - 1) / (8 * TPW);
    gemm1_kernel<<<gblocks, 256, 0, stream>>>(xb, nbr, W1, b1, h1, sum1, sq1, E);

    // 5) BN1 fold
    bn_finalize_kernel<<<1, 64, 0, stream>>>(sum1, sq1, g1, beta1, a1c, c1c, invE);

    // 6) layer 2 GEMM (WMMA, fused BN1+ReLU) + stats
    gemm2_kernel<<<gblocks, 256, 0, stream>>>(h1, W2, b2, a1c, c1c, h2, sum2, sq2, E);

    // 7) BN2 fold
    bn_finalize_kernel<<<1, 64, 0, stream>>>(sum2, sq2, g2, beta2, a2c, c2c, invE);

    // 8) BN2+ReLU + segment-max by neighbor row
    scatter_max_kernel<<<(E + 255) / 256, 256, 0, stream>>>(h2, nbr, a2c, c2c,
                                                            (unsigned*)d_out, E);
}
